// NodeExtractionGraphConvolutional_3135326126153
// MI455X (gfx1250) — matrix operand order repaired and checked by reading, not compiled, not run on hardware
//
#include <hip/hip_runtime.h>

// CDNA5 / gfx1250 implementation of NodeExtractionGraphConvolutional forward.
// All GEMMs use v_wmma_f32_16x16x32_bf16 (wave32). Weights are pre-converted to
// transposed ([n][k]) zero-padded bf16 so B fragments are single 32B vector loads.
// The layer-1 edge MLP is dead code in the reference (edge_out discarded) and is skipped.
// Stages use dual N-tile accumulators + B double-buffering to overlap L2 latency
// with the multicycle WMMAs.

typedef __attribute__((ext_vector_type(16))) __bf16 v16bf;
typedef __attribute__((ext_vector_type(8)))  __bf16 v8bf;
typedef __attribute__((ext_vector_type(4)))  __bf16 v4bf;
typedef __attribute__((ext_vector_type(8)))  float  v8f;

#define WAVES 4
#define TPB   (WAVES * 32)

// ---- bf16 weight blob offsets (elements); all matrices stored transposed [n][k] ----
static constexpr size_t OFF_W0  = 0;                       // 192 x 384
static constexpr size_t OFF_W1  = OFF_W0  + 192ull * 384;  //  96 x 192
static constexpr size_t OFF_W2  = OFF_W1  +  96ull * 192;  //  48 x  96
static constexpr size_t OFF_W3  = OFF_W2  +  48ull *  96;  //  32 x  64  (K padded 48->64)
static constexpr size_t OFF_W4  = OFF_W3  +  32ull *  64;  //  64 x  32
static constexpr size_t OFF_W5  = OFF_W4  +  64ull *  32;  // 128 x  64
static constexpr size_t OFF_WN0 = OFF_W5  + 128ull *  64;  // 128 x 256
static constexpr size_t OFF_WN1 = OFF_WN0 + 128ull * 256;  // 128 x 256
static constexpr size_t OFF_WH  = OFF_WN1 + 128ull * 256;  // per t: 4x(128x128) + 176x128
static constexpr size_t WH_T    = 4ull * 128 * 128 + 176ull * 128;
static constexpr size_t WBF_TOTAL = OFF_WH + 2 * WH_T;

// ---- WMMA fragment loaders (contiguous k order, same in A and B) ----
// A (16x32 bf16): lane<16 -> elems 0..15 = K0..15 ; lane>=16 -> K16..31
__device__ __forceinline__ v16bf lds_load_A(const __bf16* act, int ld, int kBase, int lane) {
  const int half = (lane >> 4) & 1;
  const int m    = lane & 15;
  const __bf16* row = act + m * ld + kBase + half * 16;
  v8bf lo = *(const v8bf*)(row);
  v8bf hi = *(const v8bf*)(row + 8);
  return __builtin_shufflevector(lo, hi, 0,1,2,3,4,5,6,7,8,9,10,11,12,13,14,15);
}

// B (32x16 bf16) from transposed weights WT[n][k]: lane&15 = N column,
// lanes 0-15 hold K kBase..kBase+15, lanes 16-31 hold kBase+16..kBase+31.
__device__ __forceinline__ v16bf glb_load_BT(const __bf16* WT, int ldk, int kBase, int nBase, int lane) {
  const __bf16* p = WT + (size_t)(nBase + (lane & 15)) * ldk + kBase + ((lane >> 4) << 4);
  return *(const v16bf*)p;
}

__device__ __forceinline__ v8f acc_init(float bv) {
  v8f a;
#pragma unroll
  for (int r = 0; r < 8; ++r) a[r] = bv;
  return a;
}

// GEMM stage: 16 x fout = (16 x finPad) x (finPad x fout), +bias, ReLU, bf16 out to LDS.
// Zero-pads output columns [fout, foutPad). Dual-nt accumulators + B prefetch.
__device__ __forceinline__ void stage_lds(const __bf16* in, int ldin,
                                          __bf16* out, int ldout,
                                          const __bf16* WT, int finPad,
                                          int fout, int foutPad,
                                          const float* bias, int lane) {
  const int ln = lane & 15;
  const int mh = (lane >> 4) << 3;   // C/D layout: M = r + 8*(lane>=16)
  int nt = 0;
  for (; nt + 32 <= fout; nt += 32) {
    v8f acc0 = acc_init(bias[nt + ln]);
    v8f acc1 = acc_init(bias[nt + 16 + ln]);
    v16bf b0 = glb_load_BT(WT, finPad, 0, nt, lane);
    v16bf b1 = glb_load_BT(WT, finPad, 0, nt + 16, lane);
    for (int kt = 0; kt < finPad; kt += 32) {
      v16bf a = lds_load_A(in, ldin, kt, lane);
      v16bf p0 = b0, p1 = b1;
      if (kt + 32 < finPad) {           // prefetch next K-step's B fragments
        b0 = glb_load_BT(WT, finPad, kt + 32, nt, lane);
        b1 = glb_load_BT(WT, finPad, kt + 32, nt + 16, lane);
      }
      acc0 = __builtin_amdgcn_wmma_f32_16x16x32_bf16(false, a, false, p0, (short)0, acc0, false, false);
      acc1 = __builtin_amdgcn_wmma_f32_16x16x32_bf16(false, a, false, p1, (short)0, acc1, false, false);
    }
#pragma unroll
    for (int r = 0; r < 8; ++r) {
      out[(r + mh) * ldout + nt + ln]      = (__bf16)fmaxf(acc0[r], 0.0f);
      out[(r + mh) * ldout + nt + 16 + ln] = (__bf16)fmaxf(acc1[r], 0.0f);
    }
  }
  for (; nt < fout; nt += 16) {
    v8f acc = acc_init(bias[nt + ln]);
    v16bf b = glb_load_BT(WT, finPad, 0, nt, lane);
    for (int kt = 0; kt < finPad; kt += 32) {
      v16bf a = lds_load_A(in, ldin, kt, lane);
      v16bf pb = b;
      if (kt + 32 < finPad) b = glb_load_BT(WT, finPad, kt + 32, nt, lane);
      acc = __builtin_amdgcn_wmma_f32_16x16x32_bf16(false, a, false, pb, (short)0, acc, false, false);
    }
#pragma unroll
    for (int r = 0; r < 8; ++r)
      out[(r + mh) * ldout + nt + ln] = (__bf16)fmaxf(acc[r], 0.0f);
  }
  const int padw = foutPad - fout;
  for (int idx = lane; idx < 16 * padw; idx += 32)
    out[(idx / padw) * ldout + fout + (idx % padw)] = (__bf16)0.0f;
}

// GEMM stage writing f32 to global. ACT: 0 none, 1 relu, 2 silu.
// Optional row predicate (atom_type selection) and ragged fout (169 under 176-wide W).
template <int ACT>
__device__ __forceinline__ void stage_glb(const __bf16* in, int ldin,
                                          float* out, int ldout,
                                          const __bf16* WT, int finPad,
                                          int fout, int foutPadW,
                                          const float* bias, int biasLen,
                                          long rowBase, long rowMax,
                                          const int* rowSel, int selVal, int lane) {
  const int ln = lane & 15;
  const int mh = (lane >> 4) << 3;
  auto store16 = [&](v8f acc, int ntv) {
    const int n = ntv + ln;
#pragma unroll
    for (int r = 0; r < 8; ++r) {
      const long row = rowBase + r + mh;
      if (row >= rowMax || n >= fout) continue;
      if (rowSel && rowSel[row] != selVal) continue;
      float v = acc[r];
      if (ACT == 1) v = fmaxf(v, 0.0f);
      if (ACT == 2) v = v / (1.0f + expf(-v));
      out[row * (long)ldout + n] = v;
    }
  };
  int nt = 0;
  for (; nt + 32 <= foutPadW; nt += 32) {
    const int n0 = nt + ln, n1 = nt + 16 + ln;
    v8f acc0 = acc_init((n0 < biasLen) ? bias[n0] : 0.0f);
    v8f acc1 = acc_init((n1 < biasLen) ? bias[n1] : 0.0f);
    v16bf b0 = glb_load_BT(WT, finPad, 0, nt, lane);
    v16bf b1 = glb_load_BT(WT, finPad, 0, nt + 16, lane);
    for (int kt = 0; kt < finPad; kt += 32) {
      v16bf a = lds_load_A(in, ldin, kt, lane);
      v16bf p0 = b0, p1 = b1;
      if (kt + 32 < finPad) {
        b0 = glb_load_BT(WT, finPad, kt + 32, nt, lane);
        b1 = glb_load_BT(WT, finPad, kt + 32, nt + 16, lane);
      }
      acc0 = __builtin_amdgcn_wmma_f32_16x16x32_bf16(false, a, false, p0, (short)0, acc0, false, false);
      acc1 = __builtin_amdgcn_wmma_f32_16x16x32_bf16(false, a, false, p1, (short)0, acc1, false, false);
    }
    store16(acc0, nt);
    store16(acc1, nt + 16);
  }
  for (; nt < foutPadW; nt += 16) {
    const int n = nt + ln;
    v8f acc = acc_init((n < biasLen) ? bias[n] : 0.0f);
    v16bf b = glb_load_BT(WT, finPad, 0, nt, lane);
    for (int kt = 0; kt < finPad; kt += 32) {
      v16bf a = lds_load_A(in, ldin, kt, lane);
      v16bf pb = b;
      if (kt + 32 < finPad) b = glb_load_BT(WT, finPad, kt + 32, nt, lane);
      acc = __builtin_amdgcn_wmma_f32_16x16x32_bf16(false, a, false, pb, (short)0, acc, false, false);
    }
    store16(acc, nt);
  }
}

// pack 4 f32 -> 4 bf16
__device__ __forceinline__ v4bf pack4(float4 x) {
  v4bf h;
  h[0] = (__bf16)x.x; h[1] = (__bf16)x.y; h[2] = (__bf16)x.z; h[3] = (__bf16)x.w;
  return h;
}

// ---------------- edge MLP (layer 0 only; layer-1 edge MLP is dead code) --------------
__global__ __launch_bounds__(TPB)
void edge_mlp_kernel(const float* __restrict__ nodes,
                     const float* __restrict__ er, const float* __restrict__ ea,
                     const int* __restrict__ src, const int* __restrict__ dst,
                     const __bf16* __restrict__ wbf,
                     const float* __restrict__ b0, const float* __restrict__ b1,
                     const float* __restrict__ b2, const float* __restrict__ b3,
                     const float* __restrict__ b4, const float* __restrict__ b5,
                     float* __restrict__ edge_out, long E) {
  extern __shared__ __bf16 smem[];
  const int lane = threadIdx.x & 31;
  const int wave = threadIdx.x >> 5;
  __bf16* actA = smem + (size_t)wave * (16 * 384 + 16 * 192);
  __bf16* actB = actA + 16 * 384;
  const long tile = (long)blockIdx.x * WAVES + wave;
  const long e0 = tile * 16;
  {  // gather h = [nodes[src], nodes[dst], er, ea] as bf16 into LDS (row-major 16x384)
    const int r = lane & 15;
    const int half = lane >> 4;
    const long e = e0 + r;
    int s = 0, d = 0;
    const bool ok = (e < E);
    if (ok) { s = src[e]; d = dst[e]; }
    for (int c = half * 192; c < half * 192 + 192; c += 4) {
      float4 x = make_float4(0.f, 0.f, 0.f, 0.f);
      if (ok) {
        if (c < 128)      x = *(const float4*)(nodes + (size_t)s * 128 + c);
        else if (c < 256) x = *(const float4*)(nodes + (size_t)d * 128 + (c - 128));
        else if (c < 320) x = *(const float4*)(er + e * 64 + (c - 256));
        else              x = *(const float4*)(ea + e * 64 + (c - 320));
      }
      *(v4bf*)(actA + r * 384 + c) = pack4(x);
    }
  }
  __syncthreads();
  stage_lds(actA, 384, actB, 192, wbf + OFF_W0, 384, 192, 192, b0, lane); __syncthreads();
  stage_lds(actB, 192, actA, 384, wbf + OFF_W1, 192,  96,  96, b1, lane); __syncthreads();
  stage_lds(actA, 384, actB, 192, wbf + OFF_W2,  96,  48,  64, b2, lane); __syncthreads();
  stage_lds(actB, 192, actA, 384, wbf + OFF_W3,  64,  32,  32, b3, lane); __syncthreads();
  stage_lds(actA, 384, actB, 192, wbf + OFF_W4,  32,  64,  64, b4, lane); __syncthreads();
  stage_glb<0>(actB, 192, edge_out, 128, wbf + OFF_W5, 64, 128, 128, b5, 128,
               e0, E, nullptr, 0, lane);
}

// ---------------- node update GEMM: nodes' = silu(msg @ Wn + bn) ----------------------
__global__ __launch_bounds__(TPB)
void node_mlp_kernel(const float* __restrict__ msg, const __bf16* __restrict__ WT,
                     const float* __restrict__ bias, float* __restrict__ nodesOut, long N) {
  extern __shared__ __bf16 smem[];
  const int lane = threadIdx.x & 31;
  const int wave = threadIdx.x >> 5;
  __bf16* act = smem + (size_t)wave * (16 * 256);
  const long tile = (long)blockIdx.x * WAVES + wave;
  const long v0 = tile * 16;
  {
    const int r = lane & 15;
    const int half = lane >> 4;
    const long v = v0 + r;
    const bool ok = (v < N);
    for (int c = half * 128; c < half * 128 + 128; c += 4) {
      float4 x = make_float4(0.f, 0.f, 0.f, 0.f);
      if (ok) x = *(const float4*)(msg + v * 256 + c);
      *(v4bf*)(act + r * 256 + c) = pack4(x);
    }
  }
  __syncthreads();
  stage_glb<2>(act, 256, nodesOut, 128, WT, 256, 128, 128, bias, 128, v0, N, nullptr, 0, lane);
}

// ---------------- head MLP (per atom type, predicated store) --------------------------
__global__ __launch_bounds__(TPB)
void head_kernel(const float* __restrict__ nodes, const __bf16* __restrict__ wh,
                 const float* __restrict__ bh0, const float* __restrict__ bh1,
                 const float* __restrict__ bh2, const float* __restrict__ bh3,
                 const float* __restrict__ bh4,
                 const int* __restrict__ atom_type, int t,
                 float* __restrict__ out, long N) {
  extern __shared__ __bf16 smem[];
  const int lane = threadIdx.x & 31;
  const int wave = threadIdx.x >> 5;
  __bf16* actA = smem + (size_t)wave * (16 * 256);
  __bf16* actB = actA + 16 * 128;
  const long tile = (long)blockIdx.x * WAVES + wave;
  const long v0 = tile * 16;
  {
    const int r = lane & 15;
    const int half = lane >> 4;
    const long v = v0 + r;
    const bool ok = (v < N);
    for (int c = half * 64; c < half * 64 + 64; c += 4) {
      float4 x = make_float4(0.f, 0.f, 0.f, 0.f);
      if (ok) x = *(const float4*)(nodes + v * 128 + c);
      *(v4bf*)(actA + r * 128 + c) = pack4(x);
    }
  }
  __syncthreads();
  stage_lds(actA, 128, actB, 128, wh + 0,     128, 128, 128, bh0, lane); __syncthreads();
  stage_lds(actB, 128, actA, 128, wh + 16384, 128, 128, 128, bh1, lane); __syncthreads();
  stage_lds(actA, 128, actB, 128, wh + 32768, 128, 128, 128, bh2, lane); __syncthreads();
  stage_lds(actB, 128, actA, 128, wh + 49152, 128, 128, 128, bh3, lane); __syncthreads();
  stage_glb<0>(actA, 128, out, 169, wh + 65536, 128, 169, 176, bh4, 169,
               v0, N, atom_type, t, lane);
}

// ---------------- elementwise helpers -------------------------------------------------
__global__ void zero_kernel(float* p, long n) {
  const long i = (long)blockIdx.x * blockDim.x + threadIdx.x;
  if (i < n) p[i] = 0.0f;
}

// src is [sK][sN] row-major f32; dst is transposed padded [dN][dK] bf16 (zeros outside).
__global__ void convT_kernel(const float* __restrict__ src, __bf16* __restrict__ dst,
                             int sK, int sN, int dK, long total) {
  const long i = (long)blockIdx.x * blockDim.x + threadIdx.x;
  if (i >= total) return;
  const int n = (int)(i / dK);
  const int k = (int)(i % dK);
  const float v = (k < sK && n < sN) ? src[(size_t)k * sN + n] : 0.0f;
  dst[i] = (__bf16)v;
}

__global__ void deg_kernel(const int* __restrict__ dst, float* __restrict__ deg, long E) {
  const long e = (long)blockIdx.x * blockDim.x + threadIdx.x;
  if (e < E) atomicAdd(&deg[dst[e]], 1.0f);
}

__global__ void invsqrt_kernel(const float* __restrict__ deg, float* __restrict__ invs, long N) {
  const long i = (long)blockIdx.x * blockDim.x + threadIdx.x;
  if (i < N) invs[i] = 1.0f / sqrtf(deg[i]);
}

// sub = [nodes[src[src[e]]], er[e], ea[e]] / deg[src[e]] ; msg[dst[e]] += sub  (one wave/edge)
__global__ void scatter_kernel(const float* __restrict__ nodes,
                               const float* __restrict__ erp, int erStride,
                               const float* __restrict__ eap, int eaStride,
                               const int* __restrict__ src, const int* __restrict__ dstI,
                               const float* __restrict__ deg,
                               float* __restrict__ msg, long E) {
  const long e = (long)blockIdx.x * (blockDim.x >> 5) + (threadIdx.x >> 5);
  const int lane = threadIdx.x & 31;
  if (e >= E) return;
  const int s  = src[e];
  const int d  = dstI[e];
  const int ss = src[s];                  // reference double-indexes: src_f[src]
  const float w = 1.0f / deg[s];
#pragma unroll
  for (int j = 0; j < 8; ++j) {
    const int c = lane + 32 * j;
    float v;
    if (c < 128)      v = nodes[(size_t)ss * 128 + c];
    else if (c < 192) v = erp[e * (long)erStride + (c - 128)];
    else              v = eap[e * (long)eaStride + (c - 192)];
    atomicAdd(&msg[(size_t)d * 256 + c], v * w);
  }
}

// msg[:, :128] += nodes * inv_sqrt ; msg *= inv_sqrt
__global__ void msgfix_kernel(float* __restrict__ msg, const float* __restrict__ nodes,
                              const float* __restrict__ invs, long N) {
  const long i = (long)blockIdx.x * blockDim.x + threadIdx.x;
  if (i >= N * 256) return;
  const long v = i >> 8;
  const int c = (int)(i & 255);
  const float iv = invs[v];
  float m = msg[i];
  if (c < 128) m += nodes[v * 128 + c] * iv;
  msg[i] = m * iv;
}

// ---------------- orchestration -------------------------------------------------------
extern "C" void kernel_launch(void* const* d_in, const int* in_sizes, int n_in,
                              void* d_out, int out_size, void* d_ws, size_t ws_size,
                              hipStream_t stream) {
  (void)n_in; (void)out_size; (void)ws_size;
  const float* node_features = (const float*)d_in[0];
  const float* edge_radial   = (const float*)d_in[1];
  const float* edge_angular  = (const float*)d_in[2];
  const int*   edge_index    = (const int*)d_in[3];
  const int*   atom_type     = (const int*)d_in[4];
  const float* We[6]; const float* be[6];
  for (int i = 0; i < 6; ++i) { We[i] = (const float*)d_in[5 + 2*i]; be[i] = (const float*)d_in[6 + 2*i]; }
  const float* Wn = (const float*)d_in[17];
  const float* bn = (const float*)d_in[18];
  const float* Wh[5]; const float* bh[5];
  for (int i = 0; i < 5; ++i) { Wh[i] = (const float*)d_in[19 + 2*i]; bh[i] = (const float*)d_in[20 + 2*i]; }
  float* out = (float*)d_out;

  const long N = in_sizes[0] / 128;
  const long E = in_sizes[1] / 64;
  const int* srcI = edge_index;
  const int* dstI = edge_index + E;

  // workspace carve-up (bf16 weights, deg, inv_sqrt, nodes1/2, msg, edge_out)
  auto align256 = [](size_t x) { return (x + 255) & ~(size_t)255; };
  char* p = (char*)d_ws;
  __bf16* wbf   = (__bf16*)p;  p += align256(WBF_TOTAL * sizeof(__bf16));
  float* deg    = (float*)p;   p += align256((size_t)N * 4);
  float* invs   = (float*)p;   p += align256((size_t)N * 4);
  float* nodes1 = (float*)p;   p += align256((size_t)N * 128 * 4);
  float* nodes2 = (float*)p;   p += align256((size_t)N * 128 * 4);
  float* msg    = (float*)p;   p += align256((size_t)N * 256 * 4);
  float* edge_out = (float*)p; p += align256((size_t)E * 128 * 4);

  const int T256 = 256;
  auto nb = [](long n, int t) { return (unsigned)((n + t - 1) / t); };

  // 1) convert all live weights to transposed padded bf16
  struct CJ { const float* s; size_t off; int sK, sN, dK, dN; };
  const CJ jobs[] = {
    { We[0],           OFF_W0, 384, 192, 384, 192 },
    { We[1],           OFF_W1, 192,  96, 192,  96 },
    { We[2],           OFF_W2,  96,  48,  96,  48 },
    { We[3],           OFF_W3,  48,  32,  64,  32 },   // K padded 48->64
    { We[4],           OFF_W4,  32,  64,  32,  64 },
    { We[5],           OFF_W5,  64, 128,  64, 128 },
    { Wn,              OFF_WN0, 256, 128, 256, 128 },
    { Wn + 256 * 128,  OFF_WN1, 256, 128, 256, 128 },
    { Wh[0],           OFF_WH +          0, 128, 128, 128, 128 },
    { Wh[1],           OFF_WH +      16384, 128, 128, 128, 128 },
    { Wh[2],           OFF_WH +      32768, 128, 128, 128, 128 },
    { Wh[3],           OFF_WH +      49152, 128, 128, 128, 128 },
    { Wh[4],           OFF_WH +      65536, 128, 169, 128, 176 },   // N padded 169->176
    { Wh[0] + 16384,   OFF_WH + WH_T +     0, 128, 128, 128, 128 },
    { Wh[1] + 16384,   OFF_WH + WH_T + 16384, 128, 128, 128, 128 },
    { Wh[2] + 16384,   OFF_WH + WH_T + 32768, 128, 128, 128, 128 },
    { Wh[3] + 16384,   OFF_WH + WH_T + 49152, 128, 128, 128, 128 },
    { Wh[4] + 128*169, OFF_WH + WH_T + 65536, 128, 169, 128, 176 },
  };
  for (const CJ& j : jobs) {
    const long total = (long)j.dN * j.dK;
    convT_kernel<<<nb(total, T256), T256, 0, stream>>>(j.s, wbf + j.off, j.sK, j.sN, j.dK, total);
  }

  // 2) degree + inv_sqrt
  zero_kernel<<<nb(N, T256), T256, 0, stream>>>(deg, N);
  deg_kernel<<<nb(E, T256), T256, 0, stream>>>(dstI, deg, E);
  invsqrt_kernel<<<nb(N, T256), T256, 0, stream>>>(deg, invs, N);

  const long etiles = (E + 15) / 16;
  const long ntiles = (N + 15) / 16;
  const unsigned eblocks = (unsigned)((etiles + WAVES - 1) / WAVES);
  const unsigned nblocks = (unsigned)((ntiles + WAVES - 1) / WAVES);
  const size_t edgeLds = (size_t)WAVES * (16 * 384 + 16 * 192) * sizeof(__bf16); // 72 KB
  const size_t nodeLds = (size_t)WAVES * (16 * 256) * sizeof(__bf16);            // 32 KB
  const size_t headLds = (size_t)WAVES * (16 * 256) * sizeof(__bf16);            // 32 KB

  // ---- layer 0 ----
  edge_mlp_kernel<<<eblocks, TPB, edgeLds, stream>>>(
      node_features, edge_radial, edge_angular, srcI, dstI, wbf,
      be[0], be[1], be[2], be[3], be[4], be[5], edge_out, E);

  zero_kernel<<<nb(N * 256, T256), T256, 0, stream>>>(msg, N * 256);
  scatter_kernel<<<(unsigned)((E + 7) / 8), 256, 0, stream>>>(
      node_features, edge_radial, 64, edge_angular, 64, srcI, dstI, deg, msg, E);
  msgfix_kernel<<<nb(N * 256, T256), T256, 0, stream>>>(msg, node_features, invs, N);
  node_mlp_kernel<<<nblocks, TPB, nodeLds, stream>>>(msg, wbf + OFF_WN0, bn, nodes1, N);

  // ---- layer 1 (its edge MLP output is discarded by the reference -> skipped) ----
  zero_kernel<<<nb(N * 256, T256), T256, 0, stream>>>(msg, N * 256);
  scatter_kernel<<<(unsigned)((E + 7) / 8), 256, 0, stream>>>(
      nodes1, edge_out, 128, edge_out + 64, 128, srcI, dstI, deg, msg, E);
  msgfix_kernel<<<nb(N * 256, T256), T256, 0, stream>>>(msg, nodes1, invs, N);
  node_mlp_kernel<<<nblocks, TPB, nodeLds, stream>>>(msg, wbf + OFF_WN1, bn + 128, nodes2, N);

  // ---- heads: compute per type, store only where atom_type matches ----
  for (int t = 0; t < 2; ++t) {
    head_kernel<<<nblocks, TPB, headLds, stream>>>(
        nodes2, wbf + OFF_WH + (size_t)t * WH_T,
        bh[0] + t * 128, bh[1] + t * 128, bh[2] + t * 128, bh[3] + t * 128, bh[4] + t * 169,
        atom_type, t, out, N);
  }
}
